// MoELayer_61942018343435
// MI455X (gfx1250) — matrix operand order repaired and checked by reading, not compiled, not run on hardware
//
#include <hip/hip_runtime.h>
#include <hip/hip_bf16.h>
#include <math.h>

#define E_   64
#define K_   6
#define H_   1024
#define F_   512
#define SF_  2048
#define T_   2048
#define CAP_ 384

typedef __bf16 bf16_t;
typedef __attribute__((ext_vector_type(16))) __bf16 v16bf;
typedef __attribute__((ext_vector_type(8)))  __bf16 v8bf;
typedef __attribute__((ext_vector_type(8)))  float  v8f;
typedef int v4i_t __attribute__((vector_size(16)));   // 4 x i32, matches builtin proto

typedef __attribute__((address_space(1))) v4i_t* gv4i_p;   // global
typedef __attribute__((address_space(3))) v4i_t* lv4i_p;   // LDS
typedef __attribute__((address_space(1))) void*  gvoid_p;
typedef __attribute__((address_space(3))) void*  lvoid_p;

#if __has_builtin(__builtin_amdgcn_global_load_async_to_lds_b128) && \
    __has_builtin(__builtin_amdgcn_s_wait_asynccnt)
#define HAVE_ASYNC_LDS 1
#else
#define HAVE_ASYNC_LDS 0
#endif

// 16-byte global->LDS copy: async (no VGPR round-trip, tracked by ASYNCcnt).
__device__ __forceinline__ void copy16_g2l(const bf16_t* g, bf16_t* l) {
#if HAVE_ASYNC_LDS
  __builtin_amdgcn_global_load_async_to_lds_b128(
      (gv4i_p)(gvoid_p)g, (lv4i_p)(lvoid_p)l, 0, 0);
#else
  *(uint4*)l = *(const uint4*)g;
#endif
}

__device__ __forceinline__ void async_copy_wait() {
#if HAVE_ASYNC_LDS
  __builtin_amdgcn_s_wait_asynccnt(0);
#endif
}

__device__ __forceinline__ float gelu_tanh(float x) {
  float x3 = x * x * x;
  return 0.5f * x * (1.0f + tanhf(0.7978845608028654f * (x + 0.044715f * x3)));
}

// ---------------- fp32 -> bf16 elementwise ----------------
__global__ void cvt_bf16_kernel(const float* __restrict__ src,
                                bf16_t* __restrict__ dst, size_t n) {
  size_t i = (size_t)blockIdx.x * blockDim.x + threadIdx.x;
  size_t stride = (size_t)gridDim.x * blockDim.x;
  for (; i < n; i += stride) dst[i] = (bf16_t)src[i];
}

// ------- transpose + convert: dst[b][c][r] = src[b][r][c] (src is [b][R][C]) -------
__global__ void tcvt_bf16_kernel(const float* __restrict__ src,
                                 bf16_t* __restrict__ dst,
                                 int R, int C, size_t n) {
  size_t i = (size_t)blockIdx.x * blockDim.x + threadIdx.x;
  size_t stride = (size_t)gridDim.x * blockDim.x;
  size_t plane = (size_t)R * C;
  for (; i < n; i += stride) {
    size_t b = i / plane;
    size_t rem = i - b * plane;
    size_t c = rem / R;
    size_t r = rem - c * R;
    dst[i] = (bf16_t)src[b * plane + r * (size_t)C + c];
  }
}

// ---------------- router: logits -> sigmoid -> biased top-K -> probs ----------------
__global__ void router_kernel(const float* __restrict__ x,
                              const float* __restrict__ rw,
                              const float* __restrict__ ebias,
                              int* __restrict__ top_idx,
                              float* __restrict__ probs) {
  int t = blockIdx.x * blockDim.x + threadIdx.x;
  if (t >= T_) return;
  const float* xr = x + (size_t)t * H_;
  float sc[E_];
#pragma unroll
  for (int e = 0; e < E_; ++e) sc[e] = 0.0f;
  for (int h = 0; h < H_; ++h) {
    float xv = xr[h];
#pragma unroll
    for (int e = 0; e < E_; ++e) sc[e] += xv * rw[(size_t)e * H_ + h];
  }
#pragma unroll
  for (int e = 0; e < E_; ++e) sc[e] = 1.0f / (1.0f + __expf(-sc[e]));  // sigmoid

  unsigned long long used = 0ull;
  int   ids[K_];
  float tops[K_];
  float sum = 0.0f;
  for (int k = 0; k < K_; ++k) {
    float best = -3.4e38f;
    int bi = 0;
    for (int e = 0; e < E_; ++e) {
      if ((used >> e) & 1ull) continue;
      float a = sc[e] + ebias[e];
      if (a > best) { best = a; bi = e; }   // strict > keeps lowest index on ties
    }
    used |= 1ull << bi;
    ids[k] = bi;
    tops[k] = sc[bi];
    sum += sc[bi];
  }
  float inv = 1.0f / (sum + 1e-20f);
  for (int k = 0; k < K_; ++k) {
    top_idx[(size_t)t * K_ + k] = ids[k];
    probs[(size_t)t * K_ + k]   = tops[k] * inv;
  }
}

// ---------------- deterministic dispatch: one wave per expert ----------------
__global__ void dispatch_kernel(const int* __restrict__ top_idx,
                                int* __restrict__ pair_slot,
                                int* __restrict__ slot_token) {
  int e = blockIdx.x;
  int lane = threadIdx.x;  // wave32
  for (int s = lane; s < CAP_; s += 32) slot_token[e * CAP_ + s] = -1;
  int base = 0;
  for (int i0 = 0; i0 < T_ * K_; i0 += 32) {
    int i = i0 + lane;
    int ei = top_idx[i];
    bool match = (ei == e);
    unsigned mask = (unsigned)__ballot(match);
    int prefix = __popc(mask & ((1u << lane) - 1u));
    if (match) {
      int pos = base + prefix;
      if (pos < CAP_) {
        int slot = e * CAP_ + pos;
        slot_token[slot] = i / K_;
        pair_slot[i] = slot;
      } else {
        pair_slot[i] = -1;  // dropped (over capacity)
      }
    }
    base += __popc(mask);
  }
}

// ---------------- gather tokens into expert slot rows (bf16, zero-fill) ----------------
__global__ void gather_kernel(const float* __restrict__ x,
                              const int* __restrict__ slot_token,
                              bf16_t* __restrict__ xs) {
  int row = blockIdx.x;
  int tid = threadIdx.x;
  int tok = slot_token[row];
  bf16_t* dst = xs + (size_t)row * H_;
  if (tok < 0) {
    for (int h = tid; h < H_; h += 256) dst[h] = (bf16_t)0.0f;
  } else {
    const float* src = x + (size_t)tok * H_;
    for (int h = tid; h < H_; h += 256) dst[h] = (bf16_t)src[h];
  }
}

// ---------------- tiled bf16 WMMA GEMM: C = op(A[MxK] * Bt[NxK]^T) ----------------
// Block tile 128(M) x 128(N); 8 waves, each 32(M)x64(N); K stepped by 32 through
// double-buffered LDS filled with async global->LDS copies (overlap with WMMA).
template <bool GELU, typename OutT>
__global__ void __launch_bounds__(256)
gemm_bf16_kernel(const bf16_t* __restrict__ Ag, const bf16_t* __restrict__ Btg,
                 OutT* __restrict__ Cg, int M, int N, int Kd,
                 size_t sA, size_t sB, size_t sC) {
  // stride 40 bf16 = 80B -> every 16-bf16 chunk is 16B-aligned
  __shared__ __attribute__((aligned(16))) bf16_t As[2][128][40];
  __shared__ __attribute__((aligned(16))) bf16_t Bs[2][128][40];

  const int e = blockIdx.z;
  const bf16_t* A  = Ag  + (size_t)e * sA;
  const bf16_t* Bt = Btg + (size_t)e * sB;
  OutT*         C  = Cg  + (size_t)e * sC;

  const int nBlock = blockIdx.x * 128;
  const int mBlock = blockIdx.y * 128;

  const int tid  = threadIdx.x;
  const int lane = tid & 31;
  const int ln   = lane & 15;
  const int hh   = lane >> 4;        // 0 / 1 (half-wave)
  const int wid  = tid >> 5;         // 0..7
  const int wm   = (wid & 3) * 32;   // wave M offset (0,32,64,96)
  const int wn   = (wid >> 2) * 64;  // wave N offset (0,64)

  // staging: 128 rows x 32 K each for A and Bt; 256 thr -> 1 row-half (16 bf16) each
  const int sr = tid >> 1, sh = (tid & 1) * 16;

  v8f acc[2][4] = {};

  // prologue: stage K tile 0 into buffer 0
  {
    const bf16_t* ga = A  + (size_t)(mBlock + sr) * Kd + sh;
    const bf16_t* gb = Bt + (size_t)(nBlock + sr) * Kd + sh;
    copy16_g2l(ga,     &As[0][sr][sh]);
    copy16_g2l(ga + 8, &As[0][sr][sh + 8]);
    copy16_g2l(gb,     &Bs[0][sr][sh]);
    copy16_g2l(gb + 8, &Bs[0][sr][sh + 8]);
  }
  async_copy_wait();
  __syncthreads();

  int cur = 0;
  for (int k0 = 0; k0 < Kd; k0 += 32) {
    // stage next K tile into the other buffer (async; overlaps with WMMA below)
    if (k0 + 32 < Kd) {
      int nxt = cur ^ 1;
      const bf16_t* ga = A  + (size_t)(mBlock + sr) * Kd + (k0 + 32) + sh;
      const bf16_t* gb = Bt + (size_t)(nBlock + sr) * Kd + (k0 + 32) + sh;
      copy16_g2l(ga,     &As[nxt][sr][sh]);
      copy16_g2l(ga + 8, &As[nxt][sr][sh + 8]);
      copy16_g2l(gb,     &Bs[nxt][sr][sh]);
      copy16_g2l(gb + 8, &Bs[nxt][sr][sh + 8]);
    }

    // Fragments per ISA 16-bit layouts:
    // A 16x32: lane<16 holds K {0..7,16..23}, lane>=16 holds K {8..15,24..31}, row M=ln
    // B 32x16: lane<16 holds K {0..7,16..23}, lane>=16 holds K {8..15,24..31}, column N=ln
    v16bf afr[2], bfr[4];
#pragma unroll
    for (int i = 0; i < 2; ++i) {
      int m = wm + i * 16 + ln;
      v8bf lo = *(const v8bf*)&As[cur][m][hh * 8];
      v8bf hi = *(const v8bf*)&As[cur][m][hh * 8 + 16];
      afr[i] = __builtin_shufflevector(lo, hi, 0, 1, 2, 3, 4, 5, 6, 7,
                                               8, 9, 10, 11, 12, 13, 14, 15);
    }
#pragma unroll
    for (int j = 0; j < 4; ++j) {
      int n = wn + j * 16 + ln;
      v8bf lo = *(const v8bf*)&Bs[cur][n][hh * 8];
      v8bf hi = *(const v8bf*)&Bs[cur][n][hh * 8 + 16];
      bfr[j] = __builtin_shufflevector(lo, hi, 0, 1, 2, 3, 4, 5, 6, 7,
                                               8, 9, 10, 11, 12, 13, 14, 15);
    }
#pragma unroll
    for (int i = 0; i < 2; ++i)
#pragma unroll
      for (int j = 0; j < 4; ++j)
        acc[i][j] = __builtin_amdgcn_wmma_f32_16x16x32_bf16(
            false, afr[i], false, bfr[j], (short)0, acc[i][j], false, false);

    async_copy_wait();   // next-tile async copies must land before buffer flip
    __syncthreads();
    cur ^= 1;
  }

  // Epilogue: C/D layout — VGPR v holds M=v (lanes 0-15) / M=v+8 (lanes 16-31), N=ln
#pragma unroll
  for (int i = 0; i < 2; ++i)
#pragma unroll
    for (int j = 0; j < 4; ++j)
#pragma unroll
      for (int v = 0; v < 8; ++v) {
        int m = mBlock + wm + i * 16 + v + hh * 8;
        int n = nBlock + wn + j * 16 + ln;
        float val = acc[i][j][v];
        if (GELU) val = gelu_tanh(val);
        C[(size_t)m * N + n] = (OutT)val;
      }
}

// ---------------- combine: out[t] += sum_k prob * y[slot] (fixed order) ----------------
__global__ void combine_kernel(const float* __restrict__ y,
                               const int* __restrict__ pair_slot,
                               const float* __restrict__ probs,
                               float* __restrict__ out) {
  int t = blockIdx.x;
  int tid = threadIdx.x;
  for (int h = tid; h < H_; h += 256) {
    float acc = out[(size_t)t * H_ + h];  // shared-expert result already there
#pragma unroll
    for (int k = 0; k < K_; ++k) {
      int slot = pair_slot[t * K_ + k];
      if (slot >= 0) acc += probs[t * K_ + k] * y[(size_t)slot * H_ + h];
    }
    out[(size_t)t * H_ + h] = acc;
  }
}

extern "C" void kernel_launch(void* const* d_in, const int* in_sizes, int n_in,
                              void* d_out, int out_size, void* d_ws, size_t ws_size,
                              hipStream_t stream) {
  (void)in_sizes; (void)n_in; (void)out_size; (void)ws_size;
  const float* x   = (const float*)d_in[0];  // (B,S,H)
  const float* rw  = (const float*)d_in[1];  // (E,H)
  const float* eb  = (const float*)d_in[2];  // (E,)
  const float* w1  = (const float*)d_in[3];  // (E,H,F)
  const float* w2  = (const float*)d_in[4];  // (E,F,H)
  const float* sw1 = (const float*)d_in[5];  // (SF,H)
  const float* sw2 = (const float*)d_in[6];  // (H,SF)
  float* out = (float*)d_out;

  size_t off = 0;
  auto carve = [&](size_t bytes) -> void* {
    void* p = (char*)d_ws + off;
    off += (bytes + 255) & ~(size_t)255;
    return p;
  };
  bf16_t* xb   = (bf16_t*)carve((size_t)T_ * H_ * 2);
  bf16_t* sw1b = (bf16_t*)carve((size_t)SF_ * H_ * 2);      // already [N][K]
  bf16_t* sw2b = (bf16_t*)carve((size_t)H_ * SF_ * 2);      // already [N][K]
  bf16_t* w1t  = (bf16_t*)carve((size_t)E_ * F_ * H_ * 2);  // [E][F][H]
  bf16_t* w2t  = (bf16_t*)carve((size_t)E_ * H_ * F_ * 2);  // [E][H][F]
  int*    topi  = (int*)carve((size_t)T_ * K_ * 4);
  float*  probs = (float*)carve((size_t)T_ * K_ * 4);
  int*    pslot = (int*)carve((size_t)T_ * K_ * 4);
  int*    stok  = (int*)carve((size_t)E_ * CAP_ * 4);
  bf16_t* xs   = (bf16_t*)carve((size_t)E_ * CAP_ * H_ * 2);
  bf16_t* hmid = (bf16_t*)carve((size_t)E_ * CAP_ * F_ * 2);
  float*  ybuf = (float*)carve((size_t)E_ * CAP_ * H_ * 4);
  bf16_t* smid = (bf16_t*)carve((size_t)T_ * SF_ * 2);

  cvt_bf16_kernel<<<2048, 256, 0, stream>>>(x, xb, (size_t)T_ * H_);
  cvt_bf16_kernel<<<2048, 256, 0, stream>>>(sw1, sw1b, (size_t)SF_ * H_);
  cvt_bf16_kernel<<<2048, 256, 0, stream>>>(sw2, sw2b, (size_t)H_ * SF_);
  tcvt_bf16_kernel<<<8192, 256, 0, stream>>>(w1, w1t, H_, F_, (size_t)E_ * F_ * H_);
  tcvt_bf16_kernel<<<8192, 256, 0, stream>>>(w2, w2t, F_, H_, (size_t)E_ * H_ * F_);

  router_kernel<<<T_ / 256, 256, 0, stream>>>(x, rw, eb, topi, probs);
  dispatch_kernel<<<E_, 32, 0, stream>>>(topi, pslot, stok);
  gather_kernel<<<E_ * CAP_, 256, 0, stream>>>(x, stok, xs);

  // expert MLP: hmid = gelu(xs @ w1), y = hmid @ w2
  gemm_bf16_kernel<true, bf16_t><<<dim3(F_ / 128, CAP_ / 128, E_), 256, 0, stream>>>(
      xs, w1t, hmid, CAP_, F_, H_,
      (size_t)CAP_ * H_, (size_t)F_ * H_, (size_t)CAP_ * F_);
  gemm_bf16_kernel<false, float><<<dim3(H_ / 128, CAP_ / 128, E_), 256, 0, stream>>>(
      hmid, w2t, ybuf, CAP_, H_, F_,
      (size_t)CAP_ * F_, (size_t)H_ * F_, (size_t)CAP_ * H_);

  // shared expert: out = gelu(x @ sw1^T) @ sw2^T
  gemm_bf16_kernel<true, bf16_t><<<dim3(SF_ / 128, T_ / 128, 1), 256, 0, stream>>>(
      xb, sw1b, smid, T_, SF_, H_, 0, 0, 0);
  gemm_bf16_kernel<false, float><<<dim3(H_ / 128, T_ / 128, 1), 256, 0, stream>>>(
      smid, sw2b, out, T_, H_, SF_, 0, 0, 0);

  combine_kernel<<<T_, 256, 0, stream>>>(ybuf, pslot, probs, out);
}
